// DKVMN_6571299963213
// MI455X (gfx1250) — matrix operand order repaired and checked by reading, not compiled, not run on hardware
//
#include <hip/hip_runtime.h>

typedef __attribute__((ext_vector_type(16))) __bf16 v16bf;
typedef __attribute__((ext_vector_type(8)))  __bf16 v8bf;    // 16B: one b128
typedef __attribute__((ext_vector_type(8)))  float  v8f;

#define B_TOT 256
#define T_TOT 256
#define DK    128
#define DV    128
#define CC    32
#define NUMQ  1000
#define BT    16            // batch rows per workgroup (WMMA M)

// ---- dynamic LDS layout (bytes) ----
#define MV_OFF   0
#define MV_SZ    (BT*CC*DV*4)          // 262144 : recurrent state Mv[16][32][128] f32
#define ARK_OFF  (MV_OFF + MV_SZ)
#define ARK_SZ   (BT*256*2)            // 8192   : bf16 [rt | kt] A-operand, 16x256
#define AV_OFF   (ARK_OFF + ARK_SZ)
#define AV_SZ    (BT*DV*2)             // 4096   : bf16 vt A-operand, 16x128
#define WL_OFF   (AV_OFF + AV_SZ)
#define WL_SZ    (BT*CC*4)             // 2048   : logits -> softmax weights
#define ET_OFF   (WL_OFF + WL_SZ)
#define ET_SZ    (BT*DV*4)             // 8192
#define AT_OFF   (ET_OFF + ET_SZ)
#define AT_SZ    (BT*DV*4)             // 8192
#define PACC_OFF (AT_OFF + AT_SZ)
#define SMEM_BYTES (PACC_OFF + BT*4)   // 292928 total (< 320KB WGP LDS)

// ---- bf16 workspace layout (elements) ----
#define WS_MK   0
#define WS_EW   4096
#define WS_AW   20480
#define WS_FW   36864
#define WS_KE   69632                  // k_emb  : 1000*128
#define WS_VE   197632                 // v_emb  : 2000*128
#define WS_END  453632

__device__ __forceinline__ __bf16 f2bf(float f) {
  union { float f; unsigned u; } x; x.f = f;
  unsigned r = (x.u + 0x7FFFu + ((x.u >> 16) & 1u)) >> 16;
  unsigned short h = (unsigned short)r;
  __bf16 b;
  __builtin_memcpy(&b, &h, 2);
  return b;
}

// branch-free fast transcendentals: v_exp_f32 + v_rcp_f32 only
__device__ __forceinline__ float fast_rcp(float x)     { return __builtin_amdgcn_rcpf(x); }
__device__ __forceinline__ float fast_sigmoid(float x) { return fast_rcp(1.0f + __expf(-x)); }
__device__ __forceinline__ float fast_tanh(float x)    { return 1.0f - 2.0f * fast_rcp(1.0f + __expf(2.0f * x)); }

__device__ __forceinline__ v8f wmma_bf16(v16bf a, v16bf b, v8f c) {
  return __builtin_amdgcn_wmma_f32_16x16x32_bf16(
      /*neg_a=*/false, a, /*neg_b=*/false, b,
      /*c_mod=*/(short)0, c, /*reuse_a=*/false, /*reuse_b=*/false);
}

// A fragment 16x32 bf16, row-major X (ld elems), rows 0..15, K window [k0, k0+32)
// ISA layout: lanes 0-15 = M, hold K 0..7 & 16..23; lanes 16-31 hold K 8..15 & 24..31
__device__ __forceinline__ v16bf load_a(const __bf16* X, int ld, int k0, int lane) {
  const int half = lane >> 4;
  const int m    = lane & 15;
  const __bf16* row = X + m * ld + k0 + half * 8;
  v16bf a;
#pragma unroll
  for (int e = 0; e < 16; ++e) {
    int k = (e & 7) + ((e >> 3) << 4);   // 0..7 -> 0..7 ; 8..15 -> 16..23
    a[e] = row[k];
  }
  return a;
}

// B fragment 32x16 bf16, row-major W[K][N] (ld = N elems), window (k0, n0)
// lanes 0-15: N=n, K k0..k0+7 & k0+16..k0+23 ; lanes 16-31: K k0+8..k0+15 & k0+24..k0+31
__device__ __forceinline__ v16bf load_b(const __bf16* W, int ld, int k0, int n0, int lane) {
  const int half = lane >> 4;
  const int n    = n0 + (lane & 15);
  const __bf16* col = W + (k0 + half * 8) * ld + n;
  v16bf b;
#pragma unroll
  for (int e = 0; e < 16; ++e) b[e] = col[((e & 7) + ((e >> 3) << 4)) * ld];
  return b;
}

// ---- one-time f32 -> bf16 conversion of weights AND embedding tables ----
__global__ void dkvmn_convert(const float* __restrict__ Mk,
                              const float* __restrict__ eW,
                              const float* __restrict__ aW,
                              const float* __restrict__ fW,
                              const float* __restrict__ k_emb,
                              const float* __restrict__ v_emb,
                              __bf16* __restrict__ ws) {
  int i = blockIdx.x * blockDim.x + threadIdx.x;
  if      (i < WS_EW)  ws[i] = f2bf(Mk[i]);
  else if (i < WS_AW)  ws[i] = f2bf(eW[i - WS_EW]);
  else if (i < WS_FW)  ws[i] = f2bf(aW[i - WS_AW]);
  else if (i < WS_KE)  ws[i] = f2bf(fW[i - WS_FW]);
  else if (i < WS_VE)  ws[i] = f2bf(k_emb[i - WS_KE]);
  else if (i < WS_END) ws[i] = f2bf(v_emb[i - WS_VE]);
}

__global__ void dkvmn_true_labels(const int* __restrict__ responses,
                                  float* __restrict__ out) {
  const int n = B_TOT * (T_TOT - 1);
  int i = blockIdx.x * blockDim.x + threadIdx.x;
  if (i < n) {
    int b  = i / (T_TOT - 1);
    int tt = i % (T_TOT - 1);
    out[n + i] = (float)responses[b * T_TOT + tt + 1];
  }
}

__global__ __launch_bounds__(256)
void dkvmn_recurrent_kernel(const int*   __restrict__ skills,
                            const int*   __restrict__ responses,
                            const float* __restrict__ Mv0,
                            const __bf16* __restrict__ ws,
                            const float* __restrict__ f_b,
                            const float* __restrict__ p_W,
                            const float* __restrict__ p_b,
                            const float* __restrict__ e_b,
                            const float* __restrict__ a_b,
                            float* __restrict__ out_pred) {
  extern __shared__ char smem[];
  float*  Mv   = (float*) (smem + MV_OFF);     // [BT][CC][DV]
  __bf16* a_rk = (__bf16*)(smem + ARK_OFF);    // [BT][256] = [rt | kt]
  __bf16* a_v  = (__bf16*)(smem + AV_OFF);     // [BT][DV]
  float*  wl   = (float*) (smem + WL_OFF);     // [BT][CC]
  float*  etS  = (float*) (smem + ET_OFF);     // [BT][DV]
  float*  atS  = (float*) (smem + AT_OFF);     // [BT][DV]
  float*  pacc = (float*) (smem + PACC_OFF);   // [BT]

  const __bf16* Mk_bf = ws + WS_MK;
  const __bf16* eW_bf = ws + WS_EW;
  const __bf16* aW_bf = ws + WS_AW;
  const __bf16* fW_bf = ws + WS_FW;
  const __bf16* kE_bf = ws + WS_KE;
  const __bf16* vE_bf = ws + WS_VE;

  const int tid  = threadIdx.x;
  const int lane = tid & 31;
  const int wave = tid >> 5;                   // 8 waves (wave32)
  const int half = lane >> 4;
  const int l16  = lane & 15;
  const int b0   = blockIdx.x * BT;
  const int n0w  = wave * 16;                  // this wave's N-tile

  const int r   = tid >> 4;                    // 0..15 : batch row for scalar stages
  const int c16 = tid & 15;
  const int d8  = c16 * 8;

  // ---- preload ALL weight B-fragments + per-lane bias scalars into registers ----
  v16bf eB[4], aB[4], mkB[4], fB[8];
  {
    const int n0m = (wave < 2) ? n0w : 0;      // keep addresses valid for all waves
#pragma unroll
    for (int i = 0; i < 4; ++i) {
      eB[i]  = load_b(eW_bf, DV, i * 32, n0w, lane);
      aB[i]  = load_b(aW_bf, DV, i * 32, n0w, lane);
      mkB[i] = load_b(Mk_bf, CC, i * 32, n0m, lane);
    }
#pragma unroll
    for (int i = 0; i < 8; ++i)
      fB[i] = load_b(fW_bf, DK, i * 32, n0w, lane);
  }
  const int   nlane = n0w + l16;
  const float ebn   = e_b[nlane];
  const float abn   = a_b[nlane];
  const float fbn   = f_b[nlane];
  const float pwn   = p_W[nlane];
  const float pb0   = p_b[0];

  // ---- init: broadcast Mv0 into all BT state copies ----
  for (int idx = tid; idx < BT * CC * DV; idx += 256)
    Mv[idx] = Mv0[idx & (CC * DV - 1)];

  // ---- software-pipelined gather: fetch step-0 embeddings into registers ----
  const int brow = b0 + r;
  v8bf kreg, vreg;
  {
    const int s    = skills[brow * T_TOT];
    const int resp = responses[brow * T_TOT];
    const int q    = s + NUMQ * (resp > -1 ? resp : 0);
    kreg = *(const v8bf*)(kE_bf + s * DK + d8);
    vreg = *(const v8bf*)(vE_bf + q * DV + d8);
  }
  __syncthreads();

#pragma unroll 1
  for (int t = 0; t < T_TOT; ++t) {
    // ---------- stage 1: commit step-t embeddings to LDS ----------
    *(v8bf*)(a_rk + r * 256 + DK + d8) = kreg;   // kt in cols 128..255
    *(v8bf*)(a_v  + r * DV + d8)       = vreg;
    __syncthreads();   // only dscnt must drain here; loadcnt stays free

    // ---------- prefetch step t+1 AFTER the barrier: a whole step to hide L2 latency ----------
    {
      const int tn   = (t + 1 < T_TOT) ? t + 1 : t;
      const int s    = skills[brow * T_TOT + tn];
      const int resp = responses[brow * T_TOT + tn];
      const int q    = s + NUMQ * (resp > -1 ? resp : 0);
      kreg = *(const v8bf*)(kE_bf + s * DK + d8);
      vreg = *(const v8bf*)(vE_bf + q * DV + d8);
    }

    // ---------- stage 2: WMMA GEMMs: logits = kt@Mk ; et/at pre-act = vt@{e_W,a_W} ----------
    {
      v8f eacc = {}; v8f aacc = {};
#pragma unroll
      for (int i = 0; i < 4; ++i) {
        v16bf av = load_a(a_v, DV, i * 32, lane);
        eacc = wmma_bf16(av, eB[i], eacc);
        aacc = wmma_bf16(av, aB[i], aacc);
      }
#pragma unroll
      for (int v = 0; v < 8; ++v) {
        int m = v + 8 * half;
        etS[m * DV + nlane] = fast_sigmoid(eacc[v] + ebn);
        atS[m * DV + nlane] = fast_tanh(aacc[v] + abn);
      }
      if (wave < 2) {                           // N = 32 -> 2 waves
        v8f lacc = {};
#pragma unroll
        for (int i = 0; i < 4; ++i) {
          v16bf ak = load_a(a_rk, 256, DK + i * 32, lane);
          lacc = wmma_bf16(ak, mkB[i], lacc);
        }
#pragma unroll
        for (int v = 0; v < 8; ++v)
          wl[(v + 8 * half) * CC + n0w + l16] = lacc[v];
      }
    }
    __syncthreads();

    // ---------- stage 3: per-row softmax over C=32 ; zero p accumulators ----------
    if (tid < BT) {
      float mx = -1e30f;
#pragma unroll
      for (int c = 0; c < CC; ++c) mx = fmaxf(mx, wl[tid * CC + c]);
      float s = 0.f;
#pragma unroll
      for (int c = 0; c < CC; ++c) {
        float e = __expf(wl[tid * CC + c] - mx);
        wl[tid * CC + c] = e;
        s += e;
      }
      float inv = fast_rcp(s);
#pragma unroll
      for (int c = 0; c < CC; ++c) wl[tid * CC + c] *= inv;
      pacc[tid] = 0.f;
    }
    __syncthreads();

    // ---------- stage 4: rt[b] = wt[b] . Mv[b]  (per-row matvec, VALU over LDS) ----------
    {
      float acc[8] = {0, 0, 0, 0, 0, 0, 0, 0};
#pragma unroll 4
      for (int c = 0; c < CC; ++c) {
        float w = wl[r * CC + c];
        const float* mv = &Mv[(r * CC + c) * DV + d8];
#pragma unroll
        for (int j = 0; j < 8; ++j) acc[j] += w * mv[j];
      }
      v8bf rtb;
#pragma unroll
      for (int j = 0; j < 8; ++j) rtb[j] = f2bf(acc[j]);
      *(v8bf*)(a_rk + r * 256 + d8) = rtb;      // rt in cols 0..127
    }
    __syncthreads();

    // ---------- stage 5: ft = tanh([rt,kt]@f_W + f_b); p partials ; Mv state update ----------
    {
      v8f facc = {};
#pragma unroll
      for (int i = 0; i < 8; ++i) {
        v16bf af = load_a(a_rk, 256, i * 32, lane);
        facc = wmma_bf16(af, fB[i], facc);
      }
#pragma unroll
      for (int v = 0; v < 8; ++v) {
        int m = v + 8 * half;
        float ft = fast_tanh(facc[v] + fbn);
        atomicAdd(&pacc[m], ft * pwn);          // ds_add_f32 reduction over 128 cols
      }
      // Mv[b][c][d] = Mv*(1 - w*e) + w*a
      float ev[8], aw[8];
#pragma unroll
      for (int j = 0; j < 8; ++j) {
        ev[j] = etS[r * DV + d8 + j];
        aw[j] = atS[r * DV + d8 + j];
      }
#pragma unroll 4
      for (int c = 0; c < CC; ++c) {
        float w = wl[r * CC + c];
        float* mv = &Mv[(r * CC + c) * DV + d8];
#pragma unroll
        for (int j = 0; j < 8; ++j) {
          float m_ = mv[j];
          mv[j] = m_ * (1.0f - w * ev[j]) + w * aw[j];
        }
      }
    }
    __syncthreads();

    // ---------- stage 6: p[b][t] = sigmoid(sum + p_b), emit pred for t < T-1 ----------
    if (tid < BT && t < T_TOT - 1) {
      float x = pacc[tid] + pb0;
      out_pred[(b0 + tid) * (T_TOT - 1) + t] = fast_sigmoid(x);
    }
  }
}

extern "C" void kernel_launch(void* const* d_in, const int* in_sizes, int n_in,
                              void* d_out, int out_size, void* d_ws, size_t ws_size,
                              hipStream_t stream) {
  const int*   skills    = (const int*)  d_in[0];
  const int*   responses = (const int*)  d_in[1];
  const float* k_emb     = (const float*)d_in[2];
  const float* v_emb     = (const float*)d_in[3];
  const float* Mk        = (const float*)d_in[4];
  const float* Mv0       = (const float*)d_in[5];
  const float* f_W       = (const float*)d_in[6];
  const float* f_b       = (const float*)d_in[7];
  const float* p_W       = (const float*)d_in[8];
  const float* p_b       = (const float*)d_in[9];
  const float* e_W       = (const float*)d_in[10];
  const float* e_b       = (const float*)d_in[11];
  const float* a_W       = (const float*)d_in[12];
  const float* a_b       = (const float*)d_in[13];
  float* out = (float*)d_out;

  __bf16* ws = (__bf16*)d_ws;
  dkvmn_convert<<<(WS_END + 255) / 256, 256, 0, stream>>>(Mk, e_W, a_W, f_W, k_emb, v_emb, ws);

  (void)hipFuncSetAttribute((const void*)dkvmn_recurrent_kernel,
                            hipFuncAttributeMaxDynamicSharedMemorySize, SMEM_BYTES);

  dkvmn_recurrent_kernel<<<B_TOT / BT, 256, SMEM_BYTES, stream>>>(
      skills, responses, Mv0, ws, f_b, p_W, p_b, e_b, a_b, out);

  dkvmn_true_labels<<<(B_TOT * (T_TOT - 1) + 255) / 256, 256, 0, stream>>>(responses, out);
}
